// GCN_Actor1_12884901888559
// MI455X (gfx1250) — matrix operand order repaired and checked by reading, not compiled, not run on hardware
//
#include <hip/hip_runtime.h>

// ---------------- types ----------------
typedef __attribute__((ext_vector_type(16))) __bf16 v16bf;
typedef __attribute__((ext_vector_type(8)))  float  v8f;

#define FD 128            // feature dim (F1 == H1 == H2 == 128)
#define WAVESZ 32

__device__ __forceinline__ unsigned short f32_to_bf16(float f) {
    unsigned int u = __float_as_uint(f);
    u += 0x7FFFu + ((u >> 16) & 1u);     // round-to-nearest-even
    return (unsigned short)(u >> 16);
}

// ---------------- degree / norm ----------------
__global__ void k_deg_init(int* __restrict__ deg, int N) {
    int i = blockIdx.x * blockDim.x + threadIdx.x;
    if (i < N) deg[i] = 1;               // self-loop
}

__global__ void k_deg_accum(const long long* __restrict__ col, int* __restrict__ deg, int E) {
    int e = blockIdx.x * blockDim.x + threadIdx.x;
    if (e < E) atomicAdd(&deg[(int)col[e]], 1);
}

__global__ void k_dinv(const int* __restrict__ deg, float* __restrict__ dinv, int N) {
    int i = blockIdx.x * blockDim.x + threadIdx.x;
    if (i < N) {
        int d = deg[i];
        dinv[i] = (d > 0) ? rsqrtf((float)d) : 0.0f;
    }
}

// ---------------- conversions ----------------
__global__ void k_to_bf16(const float* __restrict__ src, unsigned short* __restrict__ dst, int n) {
    int i = blockIdx.x * blockDim.x + threadIdx.x;
    if (i < n) dst[i] = f32_to_bf16(src[i]);
}

// Wt[n*128 + k] = bf16(W[k*128 + n])   (so B K-pairs are contiguous per output column)
__global__ void k_wt_bf16(const float* __restrict__ W, unsigned short* __restrict__ Wt) {
    int idx = blockIdx.x * blockDim.x + threadIdx.x;
    if (idx >= FD * FD) return;
    int n = idx >> 7, k = idx & (FD - 1);
    Wt[n * FD + k] = f32_to_bf16(W[k * FD + n]);
}

__global__ void k_zero4(float4* __restrict__ p, int n4) {
    int i = blockIdx.x * blockDim.x + threadIdx.x;
    if (i < n4) p[i] = make_float4(0.f, 0.f, 0.f, 0.f);
}

// ---------------- WMMA GEMM: C[M,128] = A[M,128] * W[128,128] ----------------
// A: bf16 row-major [M,128].  Bt: bf16 [128,128], Bt[n][k] = W[k][n].
// One wave computes a 16-row x 128-col strip: 8 output tiles, 4 k-steps of 32.
__global__ void __launch_bounds__(256)
k_gemm_bf16_wmma(const unsigned short* __restrict__ Abf,
                 const unsigned short* __restrict__ Bt,
                 float* __restrict__ C, int M) {
    const int wavesPerBlk = blockDim.x / WAVESZ;
    const int strip = blockIdx.x * wavesPerBlk + (threadIdx.x / WAVESZ);
    const int nStrips = (M + 15) / 16;
    if (strip >= nStrips) return;
    const int lane = threadIdx.x & (WAVESZ - 1);
    const int half = lane >> 4;          // 0: lanes 0-15, 1: lanes 16-31
    const int l15  = lane & 15;

    int arow = strip * 16 + l15;         // A row held by this lane (both halves: M = 0..15)
    if (arow > M - 1) arow = M - 1;      // clamp (M is a multiple of 16 here; safety only)
    const unsigned int* aPairs = (const unsigned int*)(Abf + (size_t)arow * FD);

    v8f acc[8] = {};

    #pragma unroll
    for (int kt = 0; kt < 4; ++kt) {
        const int kbase = kt * 32;
        // ---- A fragment: 16x32 bf16.  VGPR v holds K pair:
        //   K = kbase + (v>>2)*16 + half*8 + (v&3)*2
        union { v16bf v; unsigned int u[8]; } a;
        #pragma unroll
        for (int v0 = 0; v0 < 8; ++v0) {
            int K = kbase + ((v0 >> 2) << 4) + (half << 3) + ((v0 & 3) << 1);
            a.u[v0] = aPairs[K >> 1];
        }
        #pragma unroll
        for (int nt = 0; nt < 8; ++nt) {
            // ---- B fragment: 32x16 bf16.  lane -> N = l15; VGPR v holds K pair:
            //      K = kbase + (v>>2)*16 + half*8 + (v&3)*2 (same as A).
            union { v16bf v; unsigned int u[8]; } b;
            const unsigned int* bPairs =
                (const unsigned int*)(Bt + (size_t)(nt * 16 + l15) * FD);
            const int kb = (kbase + (half << 3)) >> 1;
            #pragma unroll
            for (int v0 = 0; v0 < 8; ++v0) b.u[v0] = bPairs[kb + ((v0 >> 2) << 3) + (v0 & 3)];

            acc[nt] = __builtin_amdgcn_wmma_f32_16x16x32_bf16(
                false, a.v, false, b.v, (short)0, acc[nt], false, false);
        }
    }

    // ---- store: C tile layout: lane -> N = l15, VGPR r -> M = r + half*8
    #pragma unroll
    for (int nt = 0; nt < 8; ++nt) {
        #pragma unroll
        for (int r = 0; r < 8; ++r) {
            int m = strip * 16 + r + (half << 3);
            if (m < M) C[(size_t)m * FD + nt * 16 + l15] = acc[nt][r];
        }
    }
}

// ---------------- edge scatter: agg[col] += lin[row] * dinv[row]*dinv[col] ----------------
// One wave per edge; each lane owns 4 consecutive features (32*4 = 128).
__global__ void __launch_bounds__(256)
k_scatter(const long long* __restrict__ rowI, const long long* __restrict__ colI,
          const float* __restrict__ dinv, const float* __restrict__ lin,
          float* __restrict__ agg, int E) {
    const int e = blockIdx.x * (blockDim.x / WAVESZ) + (threadIdx.x / WAVESZ);
    if (e >= E) return;
    const int lane = threadIdx.x & (WAVESZ - 1);
    const int r = (int)rowI[e];
    const int c = (int)colI[e];
    const float nrm = dinv[r] * dinv[c];
    float4 v = ((const float4*)(lin + (size_t)r * FD))[lane];
    float* dst = agg + (size_t)c * FD + lane * 4;
    unsafeAtomicAdd(dst + 0, v.x * nrm);
    unsafeAtomicAdd(dst + 1, v.y * nrm);
    unsafeAtomicAdd(dst + 2, v.z * nrm);
    unsafeAtomicAdd(dst + 3, v.w * nrm);
}

// ---------------- h = tanh(agg + lin*dinv^2 (self loop) + bias) ----------------
__global__ void k_combine(const float* __restrict__ agg, const float* __restrict__ lin,
                          const float* __restrict__ dinv, const float* __restrict__ bias,
                          float* __restrict__ out, int total) {
    int idx = blockIdx.x * blockDim.x + threadIdx.x;
    if (idx >= total) return;
    int i = idx >> 7, j = idx & (FD - 1);
    float d = dinv[i];
    out[idx] = tanhf(agg[idx] + lin[idx] * d * d + bias[j]);
}

// ---------------- head: logits[i] = dot(h[i], Wc) + bc (one wave per node) ----------------
__global__ void __launch_bounds__(256)
k_logits(const float* __restrict__ H, const float* __restrict__ Wc,
         const float* __restrict__ bc, float* __restrict__ logits, int N) {
    const int i = blockIdx.x * (blockDim.x / WAVESZ) + (threadIdx.x / WAVESZ);
    if (i >= N) return;
    const int lane = threadIdx.x & (WAVESZ - 1);
    float4 a = ((const float4*)(H + (size_t)i * FD))[lane];
    float4 w = ((const float4*)Wc)[lane];
    float s = a.x * w.x + a.y * w.y + a.z * w.z + a.w * w.w;
    #pragma unroll
    for (int off = 16; off > 0; off >>= 1) s += __shfl_down(s, off, WAVESZ);
    if (lane == 0) logits[i] = s + bc[0];
}

// ---------------- softmax over N ----------------
__global__ void k_max_partial(const float* __restrict__ x, float* __restrict__ part, int n) {
    __shared__ float s[256];
    float m = -3.402823466e38f;
    for (int i = blockIdx.x * blockDim.x + threadIdx.x; i < n; i += gridDim.x * blockDim.x)
        m = fmaxf(m, x[i]);
    s[threadIdx.x] = m; __syncthreads();
    for (int o = 128; o > 0; o >>= 1) {
        if (threadIdx.x < o) s[threadIdx.x] = fmaxf(s[threadIdx.x], s[threadIdx.x + o]);
        __syncthreads();
    }
    if (threadIdx.x == 0) part[blockIdx.x] = s[0];
}

__global__ void k_reduce_final(const float* __restrict__ part, float* __restrict__ outv,
                               int n, int isMax) {
    __shared__ float s[256];
    float acc = isMax ? -3.402823466e38f : 0.0f;
    for (int i = threadIdx.x; i < n; i += 256)
        acc = isMax ? fmaxf(acc, part[i]) : (acc + part[i]);
    s[threadIdx.x] = acc; __syncthreads();
    for (int o = 128; o > 0; o >>= 1) {
        if (threadIdx.x < o)
            s[threadIdx.x] = isMax ? fmaxf(s[threadIdx.x], s[threadIdx.x + o])
                                   : (s[threadIdx.x] + s[threadIdx.x + o]);
        __syncthreads();
    }
    if (threadIdx.x == 0) outv[0] = s[0];
}

__global__ void k_exp_partial(const float* __restrict__ logit, const float* __restrict__ maxv,
                              float* __restrict__ y, float* __restrict__ part, int n) {
    __shared__ float s[256];
    const float m = maxv[0];
    float acc = 0.0f;
    for (int i = blockIdx.x * blockDim.x + threadIdx.x; i < n; i += gridDim.x * blockDim.x) {
        float e = expf(logit[i] - m);
        y[i] = e;
        acc += e;
    }
    s[threadIdx.x] = acc; __syncthreads();
    for (int o = 128; o > 0; o >>= 1) {
        if (threadIdx.x < o) s[threadIdx.x] += s[threadIdx.x + o];
        __syncthreads();
    }
    if (threadIdx.x == 0) part[blockIdx.x] = s[0];
}

__global__ void k_normalize(float* __restrict__ y, const float* __restrict__ sumv, int n) {
    int i = blockIdx.x * blockDim.x + threadIdx.x;
    if (i < n) y[i] = y[i] / sumv[0];
}

// ---------------- launcher ----------------
extern "C" void kernel_launch(void* const* d_in, const int* in_sizes, int n_in,
                              void* d_out, int out_size, void* d_ws, size_t ws_size,
                              hipStream_t stream) {
    (void)n_in; (void)ws_size;
    const float*     x  = (const float*)d_in[0];
    const long long* ei = (const long long*)d_in[1];   // [2,E] int64
    const float*     W1 = (const float*)d_in[2];
    const float*     b1 = (const float*)d_in[3];
    const float*     W2 = (const float*)d_in[4];
    const float*     b2 = (const float*)d_in[5];
    const float*     Wc = (const float*)d_in[6];
    const float*     bc = (const float*)d_in[7];
    float* out = (float*)d_out;

    const int N = in_sizes[0] / FD;
    const int E = in_sizes[1] / 2;
    const long long total = (long long)N * FD;

    // workspace carve
    char* p = (char*)d_ws;
    auto carve = [&](size_t bytes) {
        char* r = p;
        p += (bytes + 255) & ~(size_t)255;
        return r;
    };
    float*          dinv   = (float*)carve((size_t)N * 4);           // aliased with int deg
    int*            deg    = (int*)dinv;
    float*          lin    = (float*)carve((size_t)total * 4);
    float*          agg    = (float*)carve((size_t)total * 4);
    float*          act    = (float*)carve((size_t)total * 4);
    unsigned short* hbf    = (unsigned short*)carve((size_t)total * 2);
    unsigned short* wt     = (unsigned short*)carve((size_t)FD * FD * 2);
    float*          logits = (float*)carve((size_t)N * 4);
    float*          part   = (float*)carve(512 * 4);
    float*          scal   = (float*)carve(2 * 4);

    const long long* rowI = ei;        // sources
    const long long* colI = ei + E;    // targets

    const int T = 256;
    const int wavesPerBlk = T / WAVESZ;                 // 8
    dim3 blk(T);
    int gN    = (N + T - 1) / T;
    int gE    = (E + T - 1) / T;
    int gTot  = (int)((total + T - 1) / T);
    int gTot4 = (int)((total / 4 + T - 1) / T);
    int gStr  = ((N + 15) / 16 + wavesPerBlk - 1) / wavesPerBlk;
    int gEdgW = (E + wavesPerBlk - 1) / wavesPerBlk;
    int gNW   = (N + wavesPerBlk - 1) / wavesPerBlk;

    // ---- degree / symmetric norm ----
    k_deg_init<<<gN, blk, 0, stream>>>(deg, N);
    k_deg_accum<<<gE, blk, 0, stream>>>(colI, deg, E);
    k_dinv<<<gN, blk, 0, stream>>>(deg, dinv, N);       // in-place int->float

    // ================= layer 1 =================
    k_to_bf16<<<gTot, blk, 0, stream>>>(x, hbf, (int)total);
    k_wt_bf16<<<(FD * FD + T - 1) / T, blk, 0, stream>>>(W1, wt);
    k_gemm_bf16_wmma<<<gStr, blk, 0, stream>>>(hbf, wt, lin, N);
    k_zero4<<<gTot4, blk, 0, stream>>>((float4*)agg, (int)(total / 4));
    k_scatter<<<gEdgW, blk, 0, stream>>>(rowI, colI, dinv, lin, agg, E);
    k_combine<<<gTot, blk, 0, stream>>>(agg, lin, dinv, b1, act, (int)total);

    // ================= layer 2 =================
    k_to_bf16<<<gTot, blk, 0, stream>>>(act, hbf, (int)total);
    k_wt_bf16<<<(FD * FD + T - 1) / T, blk, 0, stream>>>(W2, wt);
    k_gemm_bf16_wmma<<<gStr, blk, 0, stream>>>(hbf, wt, lin, N);
    k_zero4<<<gTot4, blk, 0, stream>>>((float4*)agg, (int)(total / 4));
    k_scatter<<<gEdgW, blk, 0, stream>>>(rowI, colI, dinv, lin, agg, E);
    k_combine<<<gTot, blk, 0, stream>>>(agg, lin, dinv, b2, act, (int)total);

    // ================= head + softmax =================
    k_logits<<<gNW, blk, 0, stream>>>(act, Wc, bc, logits, N);
    k_max_partial<<<512, blk, 0, stream>>>(logits, part, N);
    k_reduce_final<<<1, blk, 0, stream>>>(part, &scal[0], 512, 1);
    k_exp_partial<<<512, blk, 0, stream>>>(logits, &scal[0], out, part, N);
    k_reduce_final<<<1, blk, 0, stream>>>(part, &scal[1], 512, 0);
    k_normalize<<<gN, blk, 0, stream>>>(out, &scal[1], N);
}
